// RNN_41377714930269
// MI455X (gfx1250) — matrix operand order repaired and checked by reading, not compiled, not run on hardware
//
#include <hip/hip_runtime.h>
#include <math.h>

typedef __attribute__((ext_vector_type(16))) _Float16 v16h;
typedef __attribute__((ext_vector_type(8)))  _Float16 v8h;
typedef __attribute__((ext_vector_type(8)))  float    v8f;

namespace {
constexpr int LSTEPS  = 2048;
constexpr int B       = 128;
constexpr int D       = 128;
constexpr int H       = 256;
constexpr int WH_S    = 264;   // padded LDS row stride (f16 elems): 132 dwords -> banks 4*row mod 64, conflict-free
constexpr int THREADS = 512;   // 16 waves of 32; wave w owns output N-tile w (16 columns)
}

#define CVT_F4(dstvec, base, f)                         \
  dstvec[(base) + 0] = (_Float16)(f).x;                 \
  dstvec[(base) + 1] = (_Float16)(f).y;                 \
  dstvec[(base) + 2] = (_Float16)(f).z;                 \
  dstvec[(base) + 3] = (_Float16)(f).w

// Load h A-fragment for k-step `ks`: two contiguous 16B chunks (K=kb..kb+7, kb+16..kb+23)
#define LOAD_HFRAG(dst, ks)  do {                                              \
    const v8h _lo = hp[(ks) * 4];                                              \
    const v8h _hi = hp[(ks) * 4 + 2];                                          \
    (dst) = __builtin_shufflevector(_lo, _hi,                                  \
              0, 1, 2, 3, 4, 5, 6, 7, 8, 9, 10, 11, 12, 13, 14, 15);          \
  } while (0)

__global__ __launch_bounds__(512)
void rnn_scan_kernel(const float* __restrict__ x,
                     const float* __restrict__ h0,
                     const float* __restrict__ Wx,
                     const float* __restrict__ Wh,
                     const float* __restrict__ bh,
                     float* __restrict__ out)
{
  // Only the recurrent state lives in LDS (8.4 KB, static).
  __shared__ _Float16 lds_h[16 * WH_S];   // [m][k] f16, padded stride

  const int tid  = threadIdx.x;
  const int lane = tid & 31;
  const int b0   = blockIdx.x << 4;      // batch-row base: 8 blocks x 16 rows

  const int lmod = lane & 15;            // A: row M; B: col N; C: col N
  const int lhi  = lane >> 4;            // lane half selects K sub-range / C row half
  const int jcol = ((tid >> 5) << 4) + lmod;  // this lane's output column (wave owns N-tile)
  const float bias = bh[jcol];

  // ---- one-time preload: h0 slice -> LDS (f16) ----
  for (int i = tid; i < 16 * H; i += THREADS) {
    const int m = i >> 8, k = i & (H - 1);
    lds_h[m * WH_S + k] = (_Float16)h0[(b0 + m) * H + k];
  }

  // ---- one-time preload: per-wave register-resident B fragments (Wx: 4, Wh: 8) ----
  v16h wxf[4];
  #pragma unroll
  for (int ks = 0; ks < 4; ++ks) {
    const int koff = ks * 32 + (lhi << 3);
    const float* bp = Wx + jcol * D + koff;       // d values koff..koff+7, koff+16..koff+23
    const float4 g0 = ((const float4*)bp)[0];
    const float4 g1 = ((const float4*)bp)[1];
    const float4 g2 = ((const float4*)(bp + 16))[0];
    const float4 g3 = ((const float4*)(bp + 16))[1];
    CVT_F4(wxf[ks], 0, g0); CVT_F4(wxf[ks], 4, g1);
    CVT_F4(wxf[ks], 8, g2); CVT_F4(wxf[ks], 12, g3);
  }
  v16h whf[8];
  #pragma unroll
  for (int ks = 0; ks < 8; ++ks) {
    const int koff = ks * 32 + (lhi << 3);
    const float* bp = Wh + jcol * H + koff;       // k values koff..koff+7, koff+16..koff+23
    const float4 g0 = ((const float4*)bp)[0];
    const float4 g1 = ((const float4*)bp)[1];
    const float4 g2 = ((const float4*)(bp + 16))[0];
    const float4 g3 = ((const float4*)(bp + 16))[1];
    CVT_F4(whf[ks], 0, g0); CVT_F4(whf[ks], 4, g1);
    CVT_F4(whf[ks], 8, g2); CVT_F4(whf[ks], 12, g3);
  }

  // per-lane base for h A-fragments: row lmod, K sub-offset by lane half
  const v8h* hp = (const v8h*)(lds_h + lmod * WH_S + (lhi << 3));

  __syncthreads();

  for (int t = 0; t < LSTEPS; ++t) {
    // ---- issue first two h A-fragment loads immediately (lds_h valid after
    //      the trailing barrier of the previous step); their latency hides
    //      under the whole x-projection phase below.
    v16h a0, a1;                     // parity-fixed double buffer (no copies)
    LOAD_HFRAG(a0, 0);
    LOAD_HFRAG(a1, 1);

    const float* xt = x + (size_t)t * (B * D);

    // prefetch next timestep's x slice for this block (global_prefetch_b8)
    if (t + 1 < LSTEPS) {
      const float* xn = x + (size_t)(t + 1) * (B * D) + (size_t)b0 * D;
      __builtin_prefetch(xn + tid * 16, 0, 0);
    }

    // two accumulators: two independent 6-deep WMMA chains
    v8f acc0 = {};
    v8f acc1 = {};

    // ---- fused input projection: acc += x_t[b0:b0+16, :] @ Wx.T  (K = D = 128) ----
    #pragma unroll
    for (int ks = 0; ks < 4; ++ks) {
      const int dbase = ks * 32 + (lhi << 3);
      const float* ap = xt + (b0 + lmod) * D + dbase;
      const float4 f0 = ((const float4*)ap)[0];
      const float4 f1 = ((const float4*)ap)[1];
      const float4 f2 = ((const float4*)(ap + 16))[0];
      const float4 f3 = ((const float4*)(ap + 16))[1];
      v16h a;
      CVT_F4(a, 0, f0); CVT_F4(a, 4, f1); CVT_F4(a, 8, f2); CVT_F4(a, 12, f3);

      if (ks & 1)
        acc1 = __builtin_amdgcn_wmma_f32_16x16x32_f16(false, a, false, wxf[ks],
                                                      (short)0, acc1, false, false);
      else
        acc0 = __builtin_amdgcn_wmma_f32_16x16x32_f16(false, a, false, wxf[ks],
                                                      (short)0, acc0, false, false);
    }

    // ---- recurrent part: acc += h_prev @ Wh.T  (K = H = 256), Wh in registers ----
    // Parity double-buffered A fragments: after WMMA(ks) frees its buffer,
    // the load for ks+2 issues; its consumer is two WMMAs later, so the
    // dependency wait drops to dscnt<=2 (one full WMMA+load of slack).
    #pragma unroll
    for (int ks = 0; ks < 8; ++ks) {
      if ((ks & 1) == 0) {
        acc0 = __builtin_amdgcn_wmma_f32_16x16x32_f16(false, a0, false, whf[ks],
                                                      (short)0, acc0, false, false);
        if (ks + 2 < 8) LOAD_HFRAG(a0, ks + 2);
      } else {
        acc1 = __builtin_amdgcn_wmma_f32_16x16x32_f16(false, a1, false, whf[ks],
                                                      (short)0, acc1, false, false);
        if (ks + 2 < 8) LOAD_HFRAG(a1, ks + 2);
      }
    }

    // ---- combine, bias, tanh in f32 ----
    float val[8];
    #pragma unroll
    for (int r = 0; r < 8; ++r) val[r] = tanhf(acc0[r] + acc1[r] + bias);

    __syncthreads();   // all waves finished reading lds_h for this step

    // LDS h-update first (so the next barrier's dscnt wait isn't queued
    // behind the global output stores), then the f32 output stores.
    #pragma unroll
    for (int r = 0; r < 8; ++r) {
      const int m = r + (lhi << 3);              // C layout: M = r + 8*(lane/16)
      lds_h[m * WH_S + jcol] = (_Float16)val[r]; // f16 h for next step
    }
    float* outt = out + (size_t)t * (B * H);
    #pragma unroll
    for (int r = 0; r < 8; ++r) {
      const int m = r + (lhi << 3);
      outt[(b0 + m) * H + jcol] = val[r];        // f32 result to global
    }
    __syncthreads();   // h fully updated before next step reads it
  }
}

extern "C" void kernel_launch(void* const* d_in, const int* in_sizes, int n_in,
                              void* d_out, int out_size, void* d_ws, size_t ws_size,
                              hipStream_t stream) {
  (void)in_sizes; (void)n_in; (void)out_size; (void)d_ws; (void)ws_size;

  const float* x  = (const float*)d_in[0];   // [L,B,D]
  const float* h0 = (const float*)d_in[1];   // [B,H]
  const float* Wx = (const float*)d_in[2];   // [H,D]
  const float* Wh = (const float*)d_in[3];   // [H,H]
  const float* bh = (const float*)d_in[4];   // [H]
  float* out = (float*)d_out;                // [L,B,H]

  rnn_scan_kernel<<<dim3(B / 16), dim3(THREADS), 0, stream>>>(
      x, h0, Wx, Wh, bh, out);
}
